// RNNPredictor_67121748902302
// MI455X (gfx1250) — matrix operand order repaired and checked by reading, not compiled, not run on hardware
//
#include <hip/hip_runtime.h>

#define HIDDEN 64
#define TSTEPS 64
#define COEF_TOTAL (HIDDEN * 5 + 1 + HIDDEN * HIDDEN) /* 4417 */
#define WAVES_PER_BLOCK 8

typedef __attribute__((ext_vector_type(16))) _Float16 v16h;
typedef __attribute__((ext_vector_type(8)))  float    v8f;

__device__ __forceinline__ float fast_tanh(float x) {
#if __has_builtin(__builtin_amdgcn_tanhf)
    return __builtin_amdgcn_tanhf(x);   // gfx1250 v_tanh_f32 (TRANS op)
#else
    x = fminf(fmaxf(x, -15.0f), 15.0f);
    float e = __expf(2.0f * x);
    return (e - 1.0f) / (e + 1.0f);
#endif
}

// One wave32 per batch element. W_hh kept resident as 8 f16 WMMA A-tiles
// (4 M-tiles x 2 K-tiles, 64 VGPRs). Per timestep: build B by broadcasting
// h across all 16 N-columns, 8x v_wmma_f32_16x16x32_f16, take column 0,
// finalize (bias + x_t*w_ih, tanh) with 2 rows per lane.
__global__ __launch_bounds__(256)
void rnn_wmma_kernel(const float* __restrict__ coef,
                     const float* __restrict__ xin,
                     float* __restrict__ out, int Btot)
{
    __shared__ __align__(16) float sh_h[WAVES_PER_BLOCK][HIDDEN];
    __shared__ __align__(16) float sh_raw[WAVES_PER_BLOCK][HIDDEN];
    __shared__ __align__(16) float sh_x[WAVES_PER_BLOCK][TSTEPS];

    const int lane = threadIdx.x & 31;
    const int wv   = threadIdx.x >> 5;
    const int b    = blockIdx.x * WAVES_PER_BLOCK + wv;
    if (b >= Btot) return;

    const float* cb   = coef + (size_t)b * COEF_TOTAL;
    const float* w_ih = cb;
    const float* w_hh = cb + HIDDEN;                    // [64][64], row k, col j
    const float* b_ih = cb + HIDDEN + HIDDEN * HIDDEN;
    const float* b_hh = b_ih + HIDDEN;
    const float* fc_w = b_hh + HIDDEN;
    const float* fc_b = fc_w + HIDDEN;
    const float* h0   = fc_b + 1;
    const float* xrow = xin + (size_t)b * TSTEPS;

    float* hbuf = sh_h[wv];
    float* rbuf = sh_raw[wv];
    float* xbuf = sh_x[wv];

    const int r0 = lane, r1 = lane + 32;

    // Per-lane small coefficients (2 rows per lane)
    float wih0  = w_ih[r0],            wih1  = w_ih[r1];
    float bias0 = b_ih[r0] + b_hh[r0], bias1 = b_ih[r1] + b_hh[r1];
    float fcw0  = fc_w[r0],            fcw1  = fc_w[r1];
    float fcb   = fc_b[0];

    hbuf[r0] = h0[r0];   hbuf[r1] = h0[r1];
    xbuf[r0] = xrow[r0]; xbuf[r1] = xrow[r1];

    // Load W_hh once, converted to f16, into 8 A-tiles (ISA 16-bit A 16x32 layout):
    // lane l holds row M = l%16; lanes 0-15: K in {0..7,16..23}; lanes 16-31: +8.
    const int hb = lane >> 4;
    const int m  = lane & 15;
    v16h A[8];
#pragma unroll
    for (int mt = 0; mt < 4; ++mt) {
        const float* wrow = w_hh + (size_t)(16 * mt + m) * HIDDEN;
#pragma unroll
        for (int kt = 0; kt < 2; ++kt) {
            v16h a;
#pragma unroll
            for (int v = 0; v < 8; ++v) {
                int j = 32 * kt + 2 * v + 8 * hb + ((v >= 4) ? 8 : 0);
                a[2 * v]     = (_Float16)wrow[j];
                a[2 * v + 1] = (_Float16)wrow[j + 1];
            }
            A[mt * 2 + kt] = a;
        }
    }

    float hn0 = 0.0f, hn1 = 0.0f;
    const float2* hp = (const float2*)hbuf;

#pragma unroll 1
    for (int t = 0; t < TSTEPS; ++t) {
        // Build B (32x16, h replicated across all 16 N columns):
        // lanes 0-15: K in {0..7,16..23} (VGPR v: K=2v,2v+1 (+8 for v>=4)); lanes 16-31: +8.
        v16h B0, B1;
#pragma unroll
        for (int v = 0; v < 8; ++v) {
            int k0 = 8 * hb + 2 * v + ((v >= 4) ? 8 : 0); // K-tile 0
            float2 p0 = hp[k0 >> 1];
            B0[2 * v]     = (_Float16)p0.x;
            B0[2 * v + 1] = (_Float16)p0.y;
            float2 p1 = hp[(k0 + 32) >> 1];        // K-tile 1
            B1[2 * v]     = (_Float16)p1.x;
            B1[2 * v + 1] = (_Float16)p1.y;
        }

        // D = W_tile x H  (chain the two K-tiles through the accumulator)
        v8f acc[4];
#pragma unroll
        for (int mt = 0; mt < 4; ++mt) {
            v8f c = {};
            c = __builtin_amdgcn_wmma_f32_16x16x32_f16(
                    false, A[mt * 2 + 0], false, B0, (short)0, c, false, false);
            c = __builtin_amdgcn_wmma_f32_16x16x32_f16(
                    false, A[mt * 2 + 1], false, B1, (short)0, c, false, false);
            acc[mt] = c;
        }

        // Column N=0 lives in lane 0 (rows v) and lane 16 (rows v+8) per tile.
        if (lane == 0 || lane == 16) {
            int rofs = (lane == 16) ? 8 : 0;
#pragma unroll
            for (int mt = 0; mt < 4; ++mt)
#pragma unroll
                for (int v = 0; v < 8; ++v)
                    rbuf[16 * mt + rofs + v] = acc[mt][v];
        }
        // Same-wave DS ops are in-order: no barrier needed.

        float xt = xbuf[t];
        hn0 = fast_tanh(rbuf[r0] + bias0 + xt * wih0);
        hn1 = fast_tanh(rbuf[r1] + bias1 + xt * wih1);
        hbuf[r0] = hn0;
        hbuf[r1] = hn1;
    }

    // pred = sum(h * fc_w) + fc_b : wave32 reduction
    float p = hn0 * fcw0 + hn1 * fcw1;
#pragma unroll
    for (int off = 16; off > 0; off >>= 1)
        p += __shfl_xor(p, off, 32);
    if (lane == 0) out[b] = p + fcb;
}

extern "C" void kernel_launch(void* const* d_in, const int* in_sizes, int n_in,
                              void* d_out, int out_size, void* d_ws, size_t ws_size,
                              hipStream_t stream) {
    (void)n_in; (void)d_ws; (void)ws_size; (void)out_size;
    const float* coef = (const float*)d_in[0];   // [B, 4417] f32
    const float* xin  = (const float*)d_in[1];   // [B, 64]   f32
    float* out = (float*)d_out;                  // [B, 1]    f32

    int B = in_sizes[1] / TSTEPS;                // 16384
    dim3 block(32 * WAVES_PER_BLOCK);
    dim3 grid((B + WAVES_PER_BLOCK - 1) / WAVES_PER_BLOCK);
    rnn_wmma_kernel<<<grid, block, 0, stream>>>(coef, xin, out, B);
}
